// ReSpropAttention_90735479095756
// MI455X (gfx1250) — matrix operand order repaired and checked by reading, not compiled, not run on hardware
//
#include <hip/hip_runtime.h>

// ---------------------------------------------------------------------------
// Fused flash-attention for gfx1250 (MI455X), wave32 + WMMA bf16, transposed
// formulation:  S^T = K @ Q^T,  O^T = V^T @ P^T.
//   B=32, S=2048, DK=64, fp32 in/out, bf16 matmuls with f32 accumulation.
// Keys sit on the WMMA M axis -> softmax reductions are intra-lane VALU plus
// a single shfl_xor(16); P^T feeds the next WMMA as a B-fragment after one
// 4-dword half-wave exchange (no LDS round-trip). Double-buffered K/V LDS.
// V fragments are preloaded before the softmax block so ~40 ALU ops hide the
// LDS latency of the PV operands.
// ---------------------------------------------------------------------------

typedef __attribute__((ext_vector_type(16))) __bf16 v16bf;
typedef __attribute__((ext_vector_type(8)))  __bf16 v8bf;
typedef __attribute__((ext_vector_type(8)))  float  v8f;

#define S_LEN   2048
#define DK_LEN  64
#define QTILE   16                    // query rows per wave
#define NWAVES  8                     // waves per workgroup
#define KCHUNK  32                    // keys per iteration
#define NIT     (S_LEN / KCHUNK)      // 64 iterations

union bf8_u { v8bf v; int i[4]; };

// per-thread register staging of one K/V chunk (global -> regs)
__device__ __forceinline__
void load_chunk_regs(const float* __restrict__ Kb, const float* __restrict__ Vb,
                     int kc, int tid, float4& ka, float4& kb2, float* vreg)
{
    // K: thread t owns 8 contiguous dims of one key row (fully coalesced)
    const int key = tid >> 3;            // 0..31
    const int d0  = (tid & 7) * 8;       // 0,8,..,56
    const float* kp = Kb + (size_t)(kc + key) * DK_LEN + d0;
    ka  = *(const float4*)(kp);
    kb2 = *(const float4*)(kp + 4);
    // V: thread t owns one dim for 8 keys (coalesced across threads),
    //    so the transposed LDS write is one packed b128 store
    const int d  = tid & 63;
    const int k0 = (tid >> 6) * 8;       // 0,8,16,24
    #pragma unroll
    for (int j = 0; j < 8; ++j)
        vreg[j] = Vb[(size_t)(kc + k0 + j) * DK_LEN + d];
}

__device__ __forceinline__
void store_chunk_lds(__bf16 (*Kl)[DK_LEN], __bf16 (*Vt)[KCHUNK],
                     int tid, const float4& ka, const float4& kb2, const float* vreg)
{
    const int key = tid >> 3;
    const int d0  = (tid & 7) * 8;
    v8bf kb;
    kb[0] = (__bf16)ka.x;  kb[1] = (__bf16)ka.y;
    kb[2] = (__bf16)ka.z;  kb[3] = (__bf16)ka.w;
    kb[4] = (__bf16)kb2.x; kb[5] = (__bf16)kb2.y;
    kb[6] = (__bf16)kb2.z; kb[7] = (__bf16)kb2.w;
    *(v8bf*)&Kl[key][d0] = kb;                       // 1x ds_store_b128

    const int d  = tid & 63;
    const int k0 = (tid >> 6) * 8;
    v8bf vb;
    #pragma unroll
    for (int j = 0; j < 8; ++j) vb[j] = (__bf16)vreg[j];
    *(v8bf*)&Vt[d][k0] = vb;                         // 1x ds_store_b128
}

// assemble a 16x32 bf16 A-fragment from an LDS row (contiguous layout:
// element e covers K = hi*16 + e)
__device__ __forceinline__
v16bf lds_a_frag(const __bf16* row, int hi)
{
    const v8bf lo = *(const v8bf*)(row + hi * 16);
    const v8bf up = *(const v8bf*)(row + hi * 16 + 8);
    v16bf a;
    #pragma unroll
    for (int e = 0; e < 8; ++e) { a[e] = lo[e]; a[8 + e] = up[e]; }
    return a;
}

__global__ __launch_bounds__(NWAVES * 32)
void flash_attn_wmma_bf16_t(const float* __restrict__ Qg,
                            const float* __restrict__ Kg,
                            const float* __restrict__ Vg,
                            float* __restrict__ Og)
{
    __shared__ __align__(64) __bf16 Kl[2][KCHUNK][DK_LEN];   // [buf][key][dim]  8 KB
    __shared__ __align__(64) __bf16 Vt[2][DK_LEN][KCHUNK];   // [buf][dim][key]  8 KB

    const int tid  = threadIdx.x;
    const int wave = tid >> 5;
    const int lane = tid & 31;
    const int n    = lane & 15;      // query column (N) / fragment row (M)
    const int hi   = lane >> 4;      // half-wave select

    const int b     = blockIdx.y;
    const int qbase = blockIdx.x * (QTILE * NWAVES) + wave * QTILE;

    const float* Kb = Kg + (size_t)b * S_LEN * DK_LEN;
    const float* Vb = Vg + (size_t)b * S_LEN * DK_LEN;

    // ---- Q^T B-fragments (32x16 bf16; lane n = query, elem e: dim hi*16+e).
    //      1/sqrt(dk) folded into the bf16 conversion.
    v16bf bQ[2];
    {
        const float* qrow = Qg + ((size_t)b * S_LEN + qbase + n) * DK_LEN;
        #pragma unroll
        for (int dc = 0; dc < 2; ++dc)
            #pragma unroll
            for (int e = 0; e < 16; ++e)
                bQ[dc][e] = (__bf16)(qrow[dc * 32 + hi * 16 + e] * 0.125f);
    }

    // ---- accumulators (O^T: lane n = query, acc[t][r] = dim t*16+8*hi+r) ---
    v8f acc[4];
    #pragma unroll
    for (int t = 0; t < 4; ++t) acc[t] = v8f{};
    float m_st = -INFINITY, l_st = 0.0f;   // per-lane (per-query) softmax state

    // ---- software pipeline: stage chunk 0 ----------------------------------
    float4 ka, kb2; float vreg[8];
    load_chunk_regs(Kb, Vb, 0, tid, ka, kb2, vreg);
    store_chunk_lds(Kl[0], Vt[0], tid, ka, kb2, vreg);
    __syncthreads();

    for (int it = 0; it < NIT; ++it) {
        const int  cur  = it & 1;
        const bool more = (it + 1) < NIT;

        if (more)
            load_chunk_regs(Kb, Vb, (it + 1) * KCHUNK, tid, ka, kb2, vreg);
        if (it + 2 < NIT && tid < 64) {          // gfx1250 global_prefetch_b8
            __builtin_prefetch(Kb + (size_t)(it + 2) * KCHUNK * DK_LEN + tid * 32, 0, 0);
            __builtin_prefetch(Vb + (size_t)(it + 2) * KCHUNK * DK_LEN + tid * 32, 0, 0);
        }

        // -- fetch all K fragments first (8x ds_load_b128 in flight) ---------
        v16bf aK[2][2];
        #pragma unroll
        for (int kt = 0; kt < 2; ++kt)
            #pragma unroll
            for (int dc = 0; dc < 2; ++dc)
                aK[kt][dc] = lds_a_frag(&Kl[cur][kt * 16 + n][dc * 32], hi);

        // -- S^T = K @ Q^T : two 16(key)x16(query) tiles ---------------------
        v8f sT[2];
        #pragma unroll
        for (int kt = 0; kt < 2; ++kt) {
            v8f c = v8f{};
            c = __builtin_amdgcn_wmma_f32_16x16x32_bf16(
                    false, aK[kt][0], false, bQ[0], (short)0, c, false, false);
            c = __builtin_amdgcn_wmma_f32_16x16x32_bf16(
                    false, aK[kt][1], false, bQ[1], (short)0, c, false, false);
            sT[kt] = c;
        }

        // -- preload V fragments; softmax ALU below hides their LDS latency --
        v16bf aV[4];
        #pragma unroll
        for (int t = 0; t < 4; ++t)
            aV[t] = lds_a_frag(&Vt[cur][t * 16 + n][0], hi);

        // -- online softmax: reduce over 16 keys held in-lane + 1 shuffle ----
        float mx = fmaxf(sT[0][0], sT[1][0]);
        #pragma unroll
        for (int r = 1; r < 8; ++r)
            mx = fmaxf(mx, fmaxf(sT[0][r], sT[1][r]));
        mx = fmaxf(mx, __shfl_xor(mx, 16, 32));

        const float mnew  = fmaxf(m_st, mx);
        const float alpha = __expf(m_st - mnew);
        m_st = mnew;

        bf8_u pm[2];                       // P^T values this lane produced
        float rs = 0.0f;
        #pragma unroll
        for (int kt = 0; kt < 2; ++kt)
            #pragma unroll
            for (int r = 0; r < 8; ++r) {
                const float p = __expf(sT[kt][r] - mnew);
                rs += p;
                pm[kt].v[r] = (__bf16)p;
            }
        rs += __shfl_xor(rs, 16, 32);
        l_st = l_st * alpha + rs;
        #pragma unroll
        for (int t = 0; t < 4; ++t) acc[t] *= alpha;

        // -- P^T B-fragment: swap 4 dwords with partner half-wave ------------
        // lane hi=0 holds keys 0-7 (kt0) & 16-23 (kt1); needs keys 0-15.
        // lane hi=1 holds keys 8-15 (kt0) & 24-31 (kt1); needs keys 16-31.
        bf8_u po[2];
        #pragma unroll
        for (int w = 0; w < 4; ++w) {
            po[0].i[w] = __shfl_xor(pm[0].i[w], 16, 32);
            po[1].i[w] = __shfl_xor(pm[1].i[w], 16, 32);
        }
        const v8bf plo = hi ? po[1].v : pm[0].v;   // elems 0-7  (keys hi*16+0..7)
        const v8bf phi = hi ? pm[1].v : po[0].v;   // elems 8-15 (keys hi*16+8..15)
        v16bf bP;
        #pragma unroll
        for (int e = 0; e < 8; ++e) { bP[e] = plo[e]; bP[8 + e] = phi[e]; }

        // -- O^T += V^T @ P^T : 4 WMMAs over the 64 output dims --------------
        #pragma unroll
        for (int t = 0; t < 4; ++t)
            acc[t] = __builtin_amdgcn_wmma_f32_16x16x32_bf16(
                         false, aV[t], false, bP, (short)0, acc[t], false, false);

        // -- commit next chunk to the alternate buffer, one barrier/iter -----
        if (more)
            store_chunk_lds(Kl[cur ^ 1], Vt[cur ^ 1], tid, ka, kb2, vreg);
        __syncthreads();
    }

    // ---- epilogue: normalize, store O row (lane owns query qbase+n) --------
    const float inv = 1.0f / l_st;
    float* orow = Og + ((size_t)b * S_LEN + qbase + n) * DK_LEN;
    #pragma unroll
    for (int t = 0; t < 4; ++t) {
        float4 o0, o1;
        o0.x = acc[t][0] * inv; o0.y = acc[t][1] * inv;
        o0.z = acc[t][2] * inv; o0.w = acc[t][3] * inv;
        o1.x = acc[t][4] * inv; o1.y = acc[t][5] * inv;
        o1.z = acc[t][6] * inv; o1.w = acc[t][7] * inv;
        *(float4*)(orow + t * 16 + 8 * hi)     = o0;
        *(float4*)(orow + t * 16 + 8 * hi + 4) = o1;
    }
}

extern "C" void kernel_launch(void* const* d_in, const int* in_sizes, int n_in,
                              void* d_out, int out_size, void* d_ws, size_t ws_size,
                              hipStream_t stream) {
    const float* Q = (const float*)d_in[0];
    const float* K = (const float*)d_in[1];
    const float* V = (const float*)d_in[2];
    float* O = (float*)d_out;

    const int Bsz = in_sizes[0] / (S_LEN * DK_LEN);   // = 32
    dim3 grid(S_LEN / (QTILE * NWAVES), Bsz);
    flash_attn_wmma_bf16_t<<<grid, NWAVES * 32, 0, stream>>>(Q, K, V, O);
}
